// Attention_6313601925220
// MI455X (gfx1250) — matrix operand order repaired and checked by reading, not compiled, not run on hardware
//
#include <hip/hip_runtime.h>
#include <hip/hip_bf16.h>

// Problem constants (from reference)
#define BATCH 2
#define NHEAD 16
#define TSEQ  2048
#define NFEAT 256
#define DVAL  64
#define BH    (BATCH*NHEAD)

// Tiling
#define MBLK  64          // query rows per workgroup (4 waves x 16 rows)
#define WAVES 4

typedef __attribute__((ext_vector_type(16))) __bf16 v16bf;
typedef __attribute__((ext_vector_type(8)))  __bf16 v8bf;
typedef __attribute__((ext_vector_type(8)))  float  v8f;

__device__ __forceinline__ unsigned short bf16_rne(float x) {
  union { float f; unsigned u; } v; v.f = x;
  unsigned u = v.u;
  u += 0x7FFFu + ((u >> 16) & 1u);      // round-to-nearest-even
  return (unsigned short)(u >> 16);
}

__device__ __forceinline__ v16bf pack16(v8bf lo, v8bf hi) {
  return __builtin_shufflevector(lo, hi, 0,1,2,3,4,5,6,7,8,9,10,11,12,13,14,15);
}

// ---------------- Kernel 1: RoPE(Q) -> bf16 QR in workspace ----------------
__global__ void rope_cast_q(const float* __restrict__ Q,
                            const float* __restrict__ freqs,
                            unsigned short* __restrict__ qr) {
  long long idx = (long long)blockIdx.x * blockDim.x + threadIdx.x;
  const long long npairs = (long long)BH * TSEQ * (NFEAT / 2);
  if (idx >= npairs) return;
  int pair      = (int)(idx % (NFEAT / 2));
  long long bt  = idx / (NFEAT / 2);
  int t         = (int)(bt % TSEQ);
  long long base = bt * NFEAT + 2 * pair;
  float f  = freqs[2 * pair];                 // pair-quantized: f[2p]==f[2p+1]
  float ph = (float)t * f;
  ph -= floorf(ph);                           // phases % 1.0
  float ang = ph * 6.28318530717958647692f;
  float s, c;
  __sincosf(ang, &s, &c);
  float qe = Q[base], qo = Q[base + 1];
  qr[base]     = bf16_rne(qe * c - qo * s);   // even: v*cos - v_odd*sin
  qr[base + 1] = bf16_rne(qo * c + qe * s);   // odd : v*cos + v_even*sin
}

// -------- Kernel 2: V -> bf16, transposed to [BH, DV, T] for B-fragments ----
__global__ void cast_v_transpose(const float* __restrict__ V,
                                 unsigned short* __restrict__ vt) {
  long long idx = (long long)blockIdx.x * blockDim.x + threadIdx.x;
  const long long n = (long long)BH * TSEQ * DVAL;
  if (idx >= n) return;
  int j        = (int)(idx % DVAL);
  long long bt = idx / DVAL;
  int t        = (int)(bt % TSEQ);
  int bh       = (int)(bt / TSEQ);
  vt[(long long)bh * DVAL * TSEQ + (long long)j * TSEQ + t] = bf16_rne(V[idx]);
}

// ---------------- Kernel 3: causal S = QR*QR^T ; O = S*V --------------------
__launch_bounds__(WAVES * 32)
__global__ void attn_wmma(const __bf16* __restrict__ qr,
                          const __bf16* __restrict__ vt,
                          float* __restrict__ out) {
  // per-wave staging tile for S (bf16), row-major [16][64] -> 8 KB total
  __shared__ unsigned short sS[WAVES][16][MBLK];

  const int tid  = threadIdx.x;
  const int wave = tid >> 5;
  const int lane = tid & 31;
  const int lm   = lane & 15;   // row (A) or col (B/D) within 16-tile
  const int hi   = lane >> 4;   // lane half selects K / M sub-group

  const int mblk = blockIdx.x;                 // query block
  const int bh   = blockIdx.y;                 // batch*head
  const int t0   = mblk * MBLK + wave * 16;    // this wave's first query row

  const __bf16* Qb = qr + (long long)bh * TSEQ * NFEAT;
  const __bf16* Vb = vt + (long long)bh * DVAL * TSEQ;

  // ---- load this wave's Q A-fragments ONCE (invariant over key loop) ----
  // A fragment 16x32: lane lm = row; VGPR0-7 <- K = kc+hi*16+0..15
  //                                 (contiguous, matches B fragments)
  v16bf aq[NFEAT / 32];                        // 8 fragments = 64 VGPRs
#pragma unroll
  for (int kf = 0; kf < NFEAT / 32; ++kf) {
    const __bf16* ap = Qb + (long long)(t0 + lm) * NFEAT + kf * 32 + hi * 16;
    aq[kf] = pack16(*(const v8bf*)ap, *(const v8bf*)(ap + 8));
  }

  v8f o[4] = {};                               // 16 x 64 output accumulator

  for (int sb = 0; sb <= mblk; ++sb) {         // skip fully-masked key blocks
    const int s0 = sb * MBLK;

    // warm caches for the next key block while this one computes
    if (sb < mblk) {
      __builtin_prefetch(Qb + (long long)(s0 + MBLK + lane) * NFEAT, 0, 1);
    }

    v8f acc[4] = {};                           // 16 x 64 score accumulator

    // ---- S[16x64] = QR[t0:t0+16, :] x QR[s0:s0+64, :]^T  (K = 256) ----
#pragma unroll
    for (int kf = 0; kf < NFEAT / 32; ++kf) {
      const int kc = kf * 32;
#pragma unroll
      for (int n = 0; n < 4; ++n) {
        // B fragment 32x16 of QR^T: B[k][n] = QR[s0+n16+lm][k]; per lane the
        // 16 K-values (hi*16..hi*16+15) are contiguous in memory -> 32B load
        const __bf16* bp =
            Qb + (long long)(s0 + n * 16 + lm) * NFEAT + kc + hi * 16;
        v16bf b = *(const v16bf*)bp;
        acc[n] = __builtin_amdgcn_wmma_f32_16x16x32_bf16(
            false, aq[kf], false, b, (short)0, acc[n], false, false);
      }
    }

    // ---- strict-causal mask + f32->bf16, stage to LDS row-major [16][64] ---
#pragma unroll
    for (int n = 0; n < 4; ++n) {
#pragma unroll
      for (int r = 0; r < 8; ++r) {
        int mrow = r + 8 * hi;                 // D layout: M = r (+8 hi-half)
        int trow = t0 + mrow;
        int scol = s0 + n * 16 + lm;           // D layout: N = lane
        float val = (scol < trow) ? acc[n][r] : 0.0f;  // tril(k=-1)
        sS[wave][mrow][n * 16 + lm] = bf16_rne(val);
      }
    }
    asm volatile("s_wait_dscnt 0x0" ::: "memory");   // LDS RAW across lanes

    // ---- O[16x64] += S[16x64] x V[s0:s0+64, 0:64]  (K = 64) ----
#pragma unroll
    for (int kc = 0; kc < MBLK; kc += 32) {
      const unsigned short* ap = &sS[wave][lm][kc + hi * 16];
      v16bf a = pack16(*(const v8bf*)ap, *(const v8bf*)(ap + 8));
#pragma unroll
      for (int n = 0; n < 4; ++n) {
        // B[k][j] = V[s0+kc+k][n*16+lm] = Vt[n*16+lm][s0+kc+k]; contiguous K
        const __bf16* bp =
            Vb + (long long)(n * 16 + lm) * TSEQ + s0 + kc + hi * 16;
        v16bf b = *(const v16bf*)bp;
        o[n] = __builtin_amdgcn_wmma_f32_16x16x32_bf16(
            false, a, false, b, (short)0, o[n], false, false);
      }
    }
  }

  // ---- store O (f32, [BH, T, DV]) from D-fragment layout ----
  float* ob = out + (long long)bh * TSEQ * DVAL;
#pragma unroll
  for (int n = 0; n < 4; ++n) {
#pragma unroll
    for (int r = 0; r < 8; ++r) {
      ob[(long long)(t0 + r + 8 * hi) * DVAL + n * 16 + lm] = o[n][r];
    }
  }
}

extern "C" void kernel_launch(void* const* d_in, const int* in_sizes, int n_in,
                              void* d_out, int out_size, void* d_ws,
                              size_t ws_size, hipStream_t stream) {
  (void)in_sizes; (void)n_in; (void)out_size; (void)ws_size;
  const float* Q     = (const float*)d_in[0];
  const float* V     = (const float*)d_in[1];
  const float* freqs = (const float*)d_in[2];
  float* out         = (float*)d_out;

  // workspace: QR bf16 [BH,T,256] (33.5 MB) then Vt bf16 [BH,64,T] (8.4 MB)
  unsigned short* qr = (unsigned short*)d_ws;
  unsigned short* vt = qr + (size_t)BH * TSEQ * NFEAT;

  {
    long long npairs = (long long)BH * TSEQ * (NFEAT / 2);
    int blocks = (int)((npairs + 255) / 256);
    rope_cast_q<<<blocks, 256, 0, stream>>>(Q, freqs, qr);
  }
  {
    long long n = (long long)BH * TSEQ * DVAL;
    int blocks = (int)((n + 255) / 256);
    cast_v_transpose<<<blocks, 256, 0, stream>>>(V, vt);
  }
  {
    dim3 grid(TSEQ / MBLK, BH);                 // 32 x 32 workgroups
    attn_wmma<<<grid, WAVES * 32, 0, stream>>>((const __bf16*)qr,
                                               (const __bf16*)vt, out);
  }
}
